// DeepDFA_16870631538895
// MI455X (gfx1250) — matrix operand order repaired and checked by reading, not compiled, not run on hardware
//
#include <hip/hip_runtime.h>

#define A_N 32
#define S_N 256
#define O_N 4
#define B_N 512
#define L_N 2048
#define ROWS 16
#define WG_THREADS 512
#define GROUPS (B_N / ROWS)

typedef __attribute__((ext_vector_type(16))) __bf16        v16bf;
typedef __attribute__((ext_vector_type(8)))  float         v8f;
typedef __attribute__((ext_vector_type(4)))  unsigned int  u32x4;
typedef __attribute__((ext_vector_type(4)))  float         f32x4;

union Frag16 { u32x4 u[2]; v16bf v; };

__device__ __forceinline__ unsigned short f32_to_bf16(float x) {
  unsigned u = __float_as_uint(x);
  u += 0x7FFFu + ((u >> 16) & 1u);           // round-to-nearest-even
  return (unsigned short)(u >> 16);
}
__device__ __forceinline__ float bf16lo_to_f32(unsigned v) { return __uint_as_float(v << 16); }
__device__ __forceinline__ float bf16hi_to_f32(unsigned v) { return __uint_as_float(v & 0xFFFF0000u); }

// ---------------------------------------------------------------------------
// Prologue 1: pre-swizzle trans_prob (f32 [32][256][256]) into bf16 WMMA
// B-fragment layout: frag(a, nc, kc): lane l holds N=16*nc+l%16, 16 bf16 =
// K=32*kc+{8h+0..7, 16+8h+0..7} (h=l/16).  Linear index:
//   idx = ((a*16 + nc)*8 + kc)*512 + lane*16 + j
// ---------------------------------------------------------------------------
__global__ void pack_P_kernel(const float* __restrict__ P, unsigned short* __restrict__ Pf) {
  int idx = blockIdx.x * blockDim.x + threadIdx.x;
  if (idx >= A_N * S_N * S_N) return;
  int j    = idx & 15;
  int lane = (idx >> 4) & 31;
  int kc   = (idx >> 9) & 7;
  int nc   = (idx >> 12) & 15;
  int a    = idx >> 16;
  int k = kc * 32 + ((j < 8) ? (8 * (lane >> 4) + j) : (16 + 8 * (lane >> 4) + j - 8));
  int n = nc * 16 + (lane & 15);
  Pf[idx] = f32_to_bf16(P[(a * S_N + k) * S_N + n]);
}

// Prologue 2: narrow actions to u8 (same [b][t] layout)
__global__ void pack_act_kernel(const int* __restrict__ act, unsigned char* __restrict__ act8) {
  int idx = blockIdx.x * blockDim.x + threadIdx.x;
  if (idx >= B_N * L_N) return;
  act8[idx] = (unsigned char)(act[idx] & (A_N - 1));
}

// ---------------------------------------------------------------------------
// Main kernel: one workgroup = 16 batch rows, 16 waves (one N-chunk each).
// Per step: mask A rows per present action, accumulate bf16 WMMAs into two
// independent f32 accumulators (breaks the WMMA->WMMA D-chain hazard so the
// matrix pipe can issue back-to-back), round state to bf16 into the other LDS
// buffer, reduce reward via ds_add_f32.
// ---------------------------------------------------------------------------
__global__ __launch_bounds__(WG_THREADS)
void dfa_scan_kernel(const unsigned short* __restrict__ Pf,
                     const unsigned char*  __restrict__ act8,
                     const float*          __restrict__ F,
                     float*                __restrict__ out) {
  __shared__ alignas(16) unsigned short Sbuf[2][ROWS][S_N];  // 16 KB, bf16 state
  __shared__ alignas(16) float racc[ROWS][O_N];              // reward accumulator

  const int g    = blockIdx.x;
  const int tid  = threadIdx.x;
  const int wave = tid >> 5;          // N-chunk index (0..15), also K-slice for reward
  const int lane = tid & 31;
  const int mrow = lane & 15;         // batch row within group covered by this lane

  // init state = one-hot(state 0) per row, zero reward accumulator
  for (int i = tid; i < ROWS * S_N; i += WG_THREADS) {
    int k = i & (S_N - 1);
    Sbuf[0][i >> 8][k] = (k == 0) ? f32_to_bf16(1.0f) : (unsigned short)0;
  }
  if (tid < ROWS * O_N) (&racc[0][0])[tid] = 0.0f;
  __syncthreads();

  // preload this lane's fin_matrix slice: K in [16*wave,16*wave+16), o-pair
  const int obase = (lane < 16) ? 0 : 2;
  float f0[16], f1[16];
  #pragma unroll
  for (int k = 0; k < 16; ++k) {
    f0[k] = F[(wave * 16 + k) * O_N + obase];
    f1[k] = F[(wave * 16 + k) * O_N + obase + 1];
  }

  const unsigned char* myact_base = act8 + (g * ROWS + mrow) * L_N;
  float* out_base = out + (size_t)(g * ROWS) * L_N * O_N;

  const int run0 = (lane < 16) ? 0 : 8;   // A-fragment K-run start for this lane half
  const int mbase = (lane < 16) ? 0 : 8;  // D-fragment M base for this lane half
  int p = 0;

  #pragma unroll 1
  for (int t = 0; t < L_N; ++t) {
    // ---- 1) this step's action per row + present-action bitmask ----
    unsigned myact = (unsigned)myact_base[t];
    unsigned m = 1u << myact;
    #pragma unroll
    for (int off = 1; off < 32; off <<= 1) m |= (unsigned)__shfl_xor((int)m, off, 32);
    unsigned sm = (unsigned)__builtin_amdgcn_readfirstlane((int)m);

    // ---- 2) load unmasked A fragments (row mrow, all K) from LDS ----
    const unsigned short* srow = &Sbuf[p][mrow][0];
    u32x4 a_lo[8], a_hi[8];
    #pragma unroll
    for (int kc = 0; kc < 8; ++kc) {
      a_lo[kc] = *(const u32x4*)(srow + kc * 32 + run0);        // K run0..run0+7
      a_hi[kc] = *(const u32x4*)(srow + kc * 32 + run0 + 16);   // K run0+16..+23
    }

    // ---- 3) accumulate over distinct present actions (two independent chains) ----
    v8f d0 = {0.f, 0.f, 0.f, 0.f, 0.f, 0.f, 0.f, 0.f};
    v8f d1 = {0.f, 0.f, 0.f, 0.f, 0.f, 0.f, 0.f, 0.f};
    while (sm) {
      int a = __builtin_ctz(sm);
      sm &= sm - 1;
      bool on = (myact == (unsigned)a);
      const u32x4* bptr =
          (const u32x4*)Pf + ((((a * 16 + wave) * 8) * 512 + lane * 16) >> 3);
      u32x4 z = {0u, 0u, 0u, 0u};
      #pragma unroll
      for (int kc = 0; kc < 8; kc += 2) {
        Frag16 av0, bv0, av1, bv1;
        av0.u[0] = on ? a_lo[kc] : z;          // row masking (dual-cndmask, co-execs)
        av0.u[1] = on ? a_hi[kc] : z;
        bv0.u[0] = bptr[kc * 64];              // global_load_b128 x2 (L2-resident P')
        bv0.u[1] = bptr[kc * 64 + 1];
        av1.u[0] = on ? a_lo[kc + 1] : z;
        av1.u[1] = on ? a_hi[kc + 1] : z;
        bv1.u[0] = bptr[(kc + 1) * 64];
        bv1.u[1] = bptr[(kc + 1) * 64 + 1];
        d0 = __builtin_amdgcn_wmma_f32_16x16x32_bf16(false, av0.v, false, bv0.v,
                                                     (short)0, d0, false, false);
        d1 = __builtin_amdgcn_wmma_f32_16x16x32_bf16(false, av1.v, false, bv1.v,
                                                     (short)0, d1, false, false);
      }
    }
    v8f d = d0 + d1;   // merge the two accumulation chains (8 v_add_f32)

    // ---- 4) round next-state to bf16 into the other buffer ----
    {
      const int ncol = wave * 16 + mrow;   // D: lane -> N column, vgpr -> M row
      #pragma unroll
      for (int r = 0; r < 8; ++r)
        Sbuf[p ^ 1][mbase + r][ncol] = f32_to_bf16(d[r]);
    }
    __syncthreads();   // barrier A: next state complete (also orders racc re-zero)

    // ---- 5) reward partial: this wave's K-slice of s_next . F ----
    {
      const unsigned short* srow2 = &Sbuf[p ^ 1][mrow][wave * 16];
      u32x4 sl = *(const u32x4*)(srow2);
      u32x4 sh = *(const u32x4*)(srow2 + 8);
      unsigned sw[8] = {sl.x, sl.y, sl.z, sl.w, sh.x, sh.y, sh.z, sh.w};
      float r0 = 0.f, r1 = 0.f;
      #pragma unroll
      for (int q = 0; q < 8; ++q) {
        float e0 = bf16lo_to_f32(sw[q]);
        float e1 = bf16hi_to_f32(sw[q]);
        r0 = fmaf(e0, f0[2 * q], r0);  r0 = fmaf(e1, f0[2 * q + 1], r0);
        r1 = fmaf(e0, f1[2 * q], r1);  r1 = fmaf(e1, f1[2 * q + 1], r1);
      }
      atomicAdd(&racc[mrow][obase], r0);        // ds_add_f32
      atomicAdd(&racc[mrow][obase + 1], r1);
    }
    __syncthreads();   // barrier B: reward reduction complete

    // ---- 6) one wave stores rewards and re-zeroes the accumulator ----
    if (wave == 15 && lane < 16) {
      f32x4 rv = *(const f32x4*)&racc[lane][0];
      *(f32x4*)(out_base + ((size_t)lane * L_N + t) * O_N) = rv;
      f32x4 z4 = {0.f, 0.f, 0.f, 0.f};
      *(f32x4*)&racc[lane][0] = z4;
    }
    p ^= 1;
  }
}

// ---------------------------------------------------------------------------
extern "C" void kernel_launch(void* const* d_in, const int* in_sizes, int n_in,
                              void* d_out, int out_size, void* d_ws, size_t ws_size,
                              hipStream_t stream) {
  const int*   action_seq = (const int*)d_in[0];    // [512][2048] int
  const float* trans_prob = (const float*)d_in[1];  // [32][256][256] f32
  const float* fin_matrix = (const float*)d_in[2];  // [256][4] f32
  float* out = (float*)d_out;                       // [512][2048][4] f32

  unsigned short* Pf   = (unsigned short*)d_ws;                       // 4 MiB bf16 fragments
  unsigned char*  act8 = (unsigned char*)((char*)d_ws + (4u << 20));  // 1 MiB packed actions

  {
    int n = A_N * S_N * S_N;
    pack_P_kernel<<<(n + 255) / 256, 256, 0, stream>>>(trans_prob, Pf);
  }
  {
    int n = B_N * L_N;
    pack_act_kernel<<<(n + 255) / 256, 256, 0, stream>>>(action_seq, act8);
  }
  dfa_scan_kernel<<<GROUPS, WG_THREADS, 0, stream>>>(Pf, act8, fin_matrix, out);
}
